// WaveBNN_81106162417902
// MI455X (gfx1250) — 8-bit matrix operand order repaired and checked by reading, compiled, not run on hardware
//
#include <hip/hip_runtime.h>
#include <hip/hip_bf16.h>

typedef int   v8i __attribute__((ext_vector_type(8)));
typedef float v2f __attribute__((ext_vector_type(2)));
typedef float v8f __attribute__((ext_vector_type(8)));

namespace {
constexpr float kEps    = 1e-5f;
constexpr int   kKPad   = 1536;
constexpr int   kKBlk   = kKPad / 64;
constexpr int   kSamp   = 32;
constexpr int   kThreads = 64;
constexpr int   kXStride = 95;
}

__device__ __forceinline__ float signf(float w) {
  return (w > 0.f) ? 1.f : ((w < 0.f) ? -1.f : 0.f);
}

__device__ __forceinline__ int orig_k(int kp) {
  int base4, off, Lp, Lp4;
  if (kp < 192)      { base4 = 0;   off = 0;   Lp = 11; Lp4 = 12; }
  else if (kp < 384) { base4 = 192; off = 176; Lp = 11; Lp4 = 12; }
  else if (kp < 768) { base4 = 384; off = 352; Lp = 22; Lp4 = 24; }
  else               { base4 = 768; off = 704; Lp = 46; Lp4 = 48; }
  const int r  = kp - base4;
  const int ch = r / Lp4;
  const int p  = r - ch * Lp4;
  return (p < Lp) ? (off + ch * Lp + p) : -1;
}

__global__ void pack_fc1(const float* __restrict__ fc1_w, char* __restrict__ pB) {
  const int kb   = blockIdx.x >> 2;
  const int nt   = blockIdx.x & 3;
  const int lane = threadIdx.x;
  const int j     = nt * 16 + (lane & 15);
  const int khalf = (lane & 16) ? 16 : 0;
  unsigned int* dst = (unsigned int*)(pB + ((kb * 4 + nt) << 10) + lane * 32);
  #pragma unroll
  for (int v = 0; v < 8; ++v) {
    const int kstart = kb * 64 + (v >> 2) * 32 + khalf + (v & 3) * 4;
    unsigned int packed = 0u;
    #pragma unroll
    for (int b = 0; b < 4; ++b) {
      const int k = orig_k(kstart + b);
      int s = 0;
      if (k >= 0) {
        const float w = fc1_w[j * 1440 + k];
        s = (w > 0.f) ? 1 : ((w < 0.f) ? -1 : 0);
      }
      packed |= ((unsigned int)(unsigned char)(char)s) << (8 * b);
    }
    dst[v] = packed;
  }
}

template <int L, int Lp, int OFF4>
__device__ __forceinline__ void conv_branch(
    const float* __restrict__ xg, const float* __restrict__ w,
    const float* __restrict__ g, const float* __restrict__ be,
    const float* __restrict__ me, const float* __restrict__ va,
    float* __restrict__ xs, char* __restrict__ Abits,
    long blockBase, int tid, int wid, int lane) {
  constexpr int Ls  = L | 1;
  constexpr int Lp4 = (Lp + 3) & ~3;

  for (int idx = tid; idx < kSamp * L; idx += kThreads) {
    const int s = idx / L;
    const int c = idx - s * L;
    xs[s * Ls + c] = xg[(blockBase + s) * L + c];
  }
  __syncthreads();

  const int  n  = lane & 15;
  const bool hi = (lane & 16) != 0;

  const float a  = g[n] * rsqrtf(va[n] + kEps);
  const float cc = be[n] - me[n] * a;

  const float s0 = a * signf(w[n * 3 + 0]);
  const float s1 = a * signf(w[n * 3 + 1]);
  const float s2 = a * signf(w[n * 3 + 2]);
  v2f B0, B1;
  B0.x = hi ? s2 : s0;  B0.y = hi ? 0.f : s1;
  B1.x = hi ? s1 : 0.f; B1.y = hi ? s2 : s0;

  v8f cbias;
  #pragma unroll
  for (int r = 0; r < 8; ++r) cbias[r] = cc;

  const float* xrow = xs + (wid * 16 + n) * Ls;
  const int    d     = hi ? 2 : 0;
  const int    srow0 = wid * 16 + (hi ? 8 : 0);
  char* dst0 = Abits + srow0 * kKPad + OFF4 + n * Lp4;

  for (int p0 = 0; p0 < Lp; p0 += 4) {
    unsigned int pk[8] = {0u, 0u, 0u, 0u, 0u, 0u, 0u, 0u};
    #pragma unroll
    for (int pp = 0; pp < 4; ++pp) {
      const int p = p0 + pp;
      if (p < Lp) {
        v2f A;
        A.x = xrow[2 * p + d + 0];
        A.y = xrow[2 * p + d + 1];
        v8f d0 = __builtin_amdgcn_wmma_f32_16x16x4_f32(
            false, A, false, B0, (short)0, cbias, false, false);
        v8f d1 = __builtin_amdgcn_wmma_f32_16x16x4_f32(
            false, A, false, B1, (short)0, cbias, false, false);
        #pragma unroll
        for (int r = 0; r < 8; ++r) {
          const int sb = __float_as_int(d0[r]) & __float_as_int(d1[r]);
          pk[r] |= ((sb < 0) ? 0xFFu : 0x01u) << (8 * pp);
        }
      }
    }
    #pragma unroll
    for (int r = 0; r < 8; ++r) {
      *(unsigned int*)(dst0 + r * kKPad + p0) = pk[r];
    }
  }
  __syncthreads();
}

__global__ __launch_bounds__(kThreads) void wavebnn_main(
    const float* __restrict__ xA3, const float* __restrict__ wA3,
    const float* __restrict__ gA3, const float* __restrict__ bA3,
    const float* __restrict__ mA3, const float* __restrict__ vA3,
    const float* __restrict__ xD3, const float* __restrict__ wD3,
    const float* __restrict__ gD3, const float* __restrict__ bD3,
    const float* __restrict__ mD3, const float* __restrict__ vD3,
    const float* __restrict__ xD2, const float* __restrict__ wD2,
    const float* __restrict__ gD2, const float* __restrict__ bD2,
    const float* __restrict__ mD2, const float* __restrict__ vD2,
    const float* __restrict__ xD1, const float* __restrict__ wD1,
    const float* __restrict__ gD1, const float* __restrict__ bD1,
    const float* __restrict__ mD1, const float* __restrict__ vD1,
    const float* __restrict__ g1, const float* __restrict__ b1,
    const float* __restrict__ m1, const float* __restrict__ v1,
    const float* __restrict__ fc2w, const float* __restrict__ fc2b,
    const char* __restrict__ pB, float* __restrict__ out) {
  extern __shared__ char smem[];
  char*  Abits = smem;
  char*  Hbits = smem + kSamp * kKPad;
  float* xs    = (float*)(smem + kSamp * kKPad + kSamp * 64);

  const int  tid  = threadIdx.x;
  const int  lane = tid & 31;
  const int  wid  = tid >> 5;
  const long blockBase = (long)blockIdx.x * kSamp;

  conv_branch<24, 11,   0>(xA3, wA3, gA3, bA3, mA3, vA3, xs, Abits, blockBase, tid, wid, lane);
  conv_branch<24, 11, 192>(xD3, wD3, gD3, bD3, mD3, vD3, xs, Abits, blockBase, tid, wid, lane);
  conv_branch<47, 22, 384>(xD2, wD2, gD2, bD2, mD2, vD2, xs, Abits, blockBase, tid, wid, lane);
  conv_branch<94, 46, 768>(xD1, wD1, gD1, bD1, mD1, vD1, xs, Abits, blockBase, tid, wid, lane);

  {
    const int   mrow  = (wid << 4) + (lane & 15);
    const char* ArowA = Abits + mrow * kKPad;
    const int   khi   = (lane & 16) ? 16 : 0;

    v8i acc0 = {0, 0, 0, 0, 0, 0, 0, 0};
    v8i acc1 = acc0, acc2 = acc0, acc3 = acc0;

    for (int kb = 0; kb < kKBlk; ++kb) {
      const char* ak = ArowA + kb * 64 + khi;
      const int4 p0 = *(const int4*)(ak + 0);
      const int4 p1 = *(const int4*)(ak + 32);
      const v8i A = {p0.x, p0.y, p0.z, p0.w, p1.x, p1.y, p1.z, p1.w};

      const char* bbase = pB + (kb << 12) + lane * 32;
      #define FC1_STEP(ACC, NT)                                               \
      {                                                                       \
        const int4 q0 = *(const int4*)(bbase + ((NT) << 10) + 0);             \
        const int4 q1 = *(const int4*)(bbase + ((NT) << 10) + 16);            \
        const v8i Bv = {q0.x, q0.y, q0.z, q0.w, q1.x, q1.y, q1.z, q1.w};      \
        ACC = __builtin_amdgcn_wmma_i32_16x16x64_iu8(true, A, true, Bv, ACC,  \
                                                     false, false);           \
      }
      FC1_STEP(acc0, 0)
      FC1_STEP(acc1, 1)
      FC1_STEP(acc2, 2)
      FC1_STEP(acc3, 3)
      #undef FC1_STEP
    }

    const int jc    = lane & 15;
    const int rbase = (wid << 4) + ((lane & 16) ? 8 : 0);
    #define BN1_STORE(ACC, NT)                                                \
    {                                                                         \
      const int   j  = (NT) * 16 + jc;                                        \
      const float a1 = g1[j] * rsqrtf(v1[j] + kEps);                          \
      const float c1 = b1[j] - m1[j] * a1;                                    \
      _Pragma("unroll")                                                       \
      for (int r = 0; r < 8; ++r) {                                           \
        const float h = fmaf((float)ACC[r], a1, c1);                          \
        Hbits[(rbase + r) * 64 + j] = (h > 0.f) ? (char)1 : (char)-1;         \
      }                                                                       \
    }
    BN1_STORE(acc0, 0)
    BN1_STORE(acc1, 1)
    BN1_STORE(acc2, 2)
    BN1_STORE(acc3, 3)
    #undef BN1_STORE
  }
  __syncthreads();

  if (tid < kSamp) {
    const char* h = Hbits + tid * 64;
    float o0 = fc2b[0], o1 = fc2b[1], o2 = fc2b[2], o3 = fc2b[3], o4 = fc2b[4];
    #pragma unroll 8
    for (int j = 0; j < 64; ++j) {
      const float hv = (float)h[j];
      o0 = fmaf(hv, fc2w[0 * 64 + j], o0);
      o1 = fmaf(hv, fc2w[1 * 64 + j], o1);
      o2 = fmaf(hv, fc2w[2 * 64 + j], o2);
      o3 = fmaf(hv, fc2w[3 * 64 + j], o3);
      o4 = fmaf(hv, fc2w[4 * 64 + j], o4);
    }
    float* op = out + (blockBase + tid) * 5;
    op[0] = o0; op[1] = o1; op[2] = o2; op[3] = o3; op[4] = o4;
  }
}

extern "C" void kernel_launch(void* const* d_in, const int* in_sizes, int n_in,
                              void* d_out, int out_size, void* d_ws, size_t ws_size,
                              hipStream_t stream) {
  const float* xA3 = (const float*)d_in[0];
  const float* wA3 = (const float*)d_in[1];
  const float* gA3 = (const float*)d_in[2];
  const float* bA3 = (const float*)d_in[3];
  const float* mA3 = (const float*)d_in[4];
  const float* vA3 = (const float*)d_in[5];
  const float* xD3 = (const float*)d_in[6];
  const float* wD3 = (const float*)d_in[7];
  const float* gD3 = (const float*)d_in[8];
  const float* bD3 = (const float*)d_in[9];
  const float* mD3 = (const float*)d_in[10];
  const float* vD3 = (const float*)d_in[11];
  const float* xD2 = (const float*)d_in[12];
  const float* wD2 = (const float*)d_in[13];
  const float* gD2 = (const float*)d_in[14];
  const float* bD2 = (const float*)d_in[15];
  const float* mD2 = (const float*)d_in[16];
  const float* vD2 = (const float*)d_in[17];
  const float* xD1 = (const float*)d_in[18];
  const float* wD1 = (const float*)d_in[19];
  const float* gD1 = (const float*)d_in[20];
  const float* bD1 = (const float*)d_in[21];
  const float* mD1 = (const float*)d_in[22];
  const float* vD1 = (const float*)d_in[23];
  const float* fc1_w = (const float*)d_in[24];
  const float* g1  = (const float*)d_in[25];
  const float* b1  = (const float*)d_in[26];
  const float* m1  = (const float*)d_in[27];
  const float* v1  = (const float*)d_in[28];
  const float* fc2w = (const float*)d_in[29];
  const float* fc2b = (const float*)d_in[30];

  char* pB = (char*)d_ws;

  const int Bn = in_sizes[0] / 24;

  pack_fc1<<<dim3(kKBlk * 4), dim3(32), 0, stream>>>(fc1_w, pB);

  const size_t lds = (size_t)kSamp * kKPad
                   + (size_t)kSamp * 64
                   + (size_t)kSamp * kXStride * sizeof(float);
  wavebnn_main<<<dim3(Bn / kSamp), dim3(kThreads), lds, stream>>>(
      xA3, wA3, gA3, bA3, mA3, vA3,
      xD3, wD3, gD3, bD3, mD3, vD3,
      xD2, wD2, gD2, bD2, mD2, vD2,
      xD1, wD1, gD1, bD1, mD1, vD1,
      g1, b1, m1, v1, fc2w, fc2b, pB, (float*)d_out);
}
